// TopologicalLayer_24215025615254
// MI455X (gfx1250) — matrix operand order repaired and checked by reading, not compiled, not run on hardware
//
#include <hip/hip_runtime.h>
#include <hip/hip_bf16.h>

// ---------------- problem constants ----------------
#define N_  3072
#define DIN_ 64
#define D_  128
#define H_  4
#define L_  2
#define S_  4
#define E_  49152

// ---------------- bf16 / WMMA types ----------------
typedef __attribute__((ext_vector_type(16))) __bf16 v16bf;
typedef __attribute__((ext_vector_type(8)))  float  v8f;

union BFU { unsigned short u; __bf16 b; };

__device__ __forceinline__ unsigned short f2bf_u(float f) {
    unsigned int u = __builtin_bit_cast(unsigned int, f);
    u += 0x7FFFu + ((u >> 16) & 1u);          // round-to-nearest-even
    return (unsigned short)(u >> 16);
}
__device__ __forceinline__ __bf16 us2bf(unsigned short u) { BFU c; c.u = u; return c.b; }
__device__ __forceinline__ __bf16 f2bf(float f)           { BFU c; c.u = f2bf_u(f); return c.b; }

// ---- fragment loaders (CDNA5 wave32 16x16x32 bf16 layouts) ----
__device__ __forceinline__ void load_a_frag(v16bf& a, const unsigned short* __restrict__ A,
                                            int lda, int row, int k0, int hf)
{
    #pragma unroll
    for (int v = 0; v < 8; ++v) {
        const int kk = ((v < 4) ? (2 * v) : (16 + 2 * (v - 4))) + 8 * hf;
        const unsigned short* ap = A + (size_t)row * lda + k0 + kk;
        a[2 * v]     = us2bf(ap[0]);
        a[2 * v + 1] = us2bf(ap[1]);
    }
}

template <bool BT, bool BF32>
__device__ __forceinline__ void load_b_frag(v16bf& b, const void* __restrict__ Bp,
                                            int ldb, int k0, int col, int hf)
{
    #pragma unroll
    for (int v = 0; v < 8; ++v) {
        const int kb = ((v < 4) ? (2 * v) : (16 + 2 * (v - 4))) + 8 * hf;
        #pragma unroll
        for (int j = 0; j < 2; ++j) {
            const int kidx = k0 + kb + j;
            if (BF32) {
                const float* Bf = (const float*)Bp;
                float f = BT ? Bf[(size_t)col * ldb + kidx]
                             : Bf[(size_t)kidx * ldb + col];
                b[2 * v + j] = f2bf(f);
            } else {
                const unsigned short* Bh = (const unsigned short*)Bp;
                unsigned short u = BT ? Bh[(size_t)col * ldb + kidx]
                                      : Bh[(size_t)kidx * ldb + col];
                b[2 * v + j] = us2bf(u);
            }
        }
    }
}

// ======================================================================
// Generic bf16 WMMA GEMM, one wave computes a 16 x (16*CT) strip.
// The A fragment is loaded once per K-step and reused across CT column
// tiles (4 accumulators) -> 4x less A traffic, 4x more WMMA per wave.
//   C[r, colOff+c] = acc(r,c) + bias[c] + resid[r,c]
// Grid: (M/16, Ncols/(16*CT)), block 32 (EXEC all ones - WMMA requirement).
// ======================================================================
template <bool BT, bool BF32, bool OUTBF, int CT>
__global__ __launch_bounds__(32)
void wmma_gemm_kernel(const unsigned short* __restrict__ A, int lda,
                      const void* __restrict__ Bp, int ldb,
                      const float* __restrict__ bias,
                      const float* __restrict__ resid, int ldres,
                      float* __restrict__ Cf, unsigned short* __restrict__ Ch,
                      int ldc, int colOff, int K)
{
    const int lane = threadIdx.x;
    const int hf   = lane >> 4;     // lane half (0/1)
    const int ln16 = lane & 15;
    const int row0 = blockIdx.x * 16;
    const int col0 = blockIdx.y * 16 * CT;

    v8f acc[CT];
    #pragma unroll
    for (int ct = 0; ct < CT; ++ct) acc[ct] = (v8f){};

    for (int k0 = 0; k0 < K; k0 += 32) {
        if (k0 + 32 < K)   // CDNA5 global_prefetch_b8 for the next A tile
            __builtin_prefetch(A + (size_t)(row0 + ln16) * lda + k0 + 32, 0, 1);

        v16bf a;
        load_a_frag(a, A, lda, row0 + ln16, k0, hf);
        #pragma unroll
        for (int ct = 0; ct < CT; ++ct) {
            v16bf b;
            load_b_frag<BT, BF32>(b, Bp, ldb, k0, col0 + ct * 16 + ln16, hf);
            acc[ct] = __builtin_amdgcn_wmma_f32_16x16x32_bf16(
                          false, a, false, b, (short)0, acc[ct], false, false);
        }
    }

    // ---- C/D layout: VGPR v, lane half hf -> row v+8*hf, col ln16 ----
    #pragma unroll
    for (int ct = 0; ct < CT; ++ct) {
        #pragma unroll
        for (int v = 0; v < 8; ++v) {
            const int r = row0 + v + 8 * hf;
            const int c = col0 + ct * 16 + ln16;
            float val = acc[ct][v];
            if (bias)  val += bias[c];
            if (resid) val += resid[(size_t)r * ldres + c];
            const int co = c + colOff;
            if (OUTBF) Ch[(size_t)r * ldc + co] = f2bf_u(val);
            else       Cf[(size_t)r * ldc + co] = val;
        }
    }
}

// ======================================================================
// QK^T with scale + precomputed (masked) bias, 16 x 64 strip per wave.
//   logits[n,m] = masked ? -3e38 : scale * sum_d Q[n,d]K[m,d] + bias[n,m]
// Grid: (N/16, N/64), block 32.
// ======================================================================
__global__ __launch_bounds__(32)
void wmma_qkt_kernel(const unsigned short* __restrict__ Q,
                     const unsigned short* __restrict__ Km,
                     const float* __restrict__ biasM,
                     float* __restrict__ logits, float scale)
{
    const int lane = threadIdx.x;
    const int hf   = lane >> 4;
    const int ln16 = lane & 15;
    const int row0 = blockIdx.x * 16;
    const int col0 = blockIdx.y * 64;

    v8f acc[4];
    #pragma unroll
    for (int ct = 0; ct < 4; ++ct) acc[ct] = (v8f){};

    #pragma unroll
    for (int k0 = 0; k0 < D_; k0 += 32) {
        v16bf a;
        load_a_frag(a, Q, D_, row0 + ln16, k0, hf);
        #pragma unroll
        for (int ct = 0; ct < 4; ++ct) {
            v16bf b;   // B(k,m) = K[m,k]  (transposed access, contiguous in k)
            load_b_frag<true, false>(b, Km, D_, k0, col0 + ct * 16 + ln16, hf);
            acc[ct] = __builtin_amdgcn_wmma_f32_16x16x32_bf16(
                          false, a, false, b, (short)0, acc[ct], false, false);
        }
    }

    #pragma unroll
    for (int ct = 0; ct < 4; ++ct) {
        #pragma unroll
        for (int v = 0; v < 8; ++v) {
            const int r = row0 + v + 8 * hf;
            const int c = col0 + ct * 16 + ln16;
            const float bm = biasM[(size_t)r * N_ + c];
            logits[(size_t)r * N_ + c] =
                (bm < -1.0e29f) ? -3.0e38f : fmaf(acc[ct][v], scale, bm);
        }
    }
}

// ======================================================================
// Row softmax with mask sentinel handling; writes bf16 P.
// Fully-masked rows (denom==0) -> all zeros (matches reference zeroing).
// One 256-thread block per row; each thread owns N/256 = 12 elements.
// ======================================================================
__global__ __launch_bounds__(256)
void softmax_kernel(const float* __restrict__ logits, unsigned short* __restrict__ P)
{
    __shared__ float red[8];
    const int row = blockIdx.x;
    const int t   = threadIdx.x;
    const float* Lr = logits + (size_t)row * N_;

    float vals[N_ / 256];
    float mx = -3.4e38f;
    #pragma unroll
    for (int i = 0; i < N_ / 256; ++i) {
        vals[i] = Lr[t + i * 256];
        mx = fmaxf(mx, vals[i]);
    }
    #pragma unroll
    for (int off = 16; off; off >>= 1) mx = fmaxf(mx, __shfl_xor(mx, off, 32));
    if ((t & 31) == 0) red[t >> 5] = mx;
    __syncthreads();
    float m = red[0];
    #pragma unroll
    for (int i = 1; i < 8; ++i) m = fmaxf(m, red[i]);
    __syncthreads();

    float s = 0.f;
    #pragma unroll
    for (int i = 0; i < N_ / 256; ++i) {
        float p = (vals[i] < -1.0e29f) ? 0.f : __expf(vals[i] - m);
        vals[i] = p;
        s += p;
    }
    #pragma unroll
    for (int off = 16; off; off >>= 1) s += __shfl_xor(s, off, 32);
    if ((t & 31) == 0) red[t >> 5] = s;
    __syncthreads();
    float tot = 0.f;
    #pragma unroll
    for (int i = 0; i < 8; ++i) tot += red[i];
    const float inv = (tot > 0.f) ? (1.0f / tot) : 0.f;

    #pragma unroll
    for (int i = 0; i < N_ / 256; ++i)
        P[(size_t)row * N_ + t + i * 256] = f2bf_u(vals[i] * inv);
}

// ======================================================================
// LayerNorm over D=128, output bf16.  One 128-thread block per row.
// ======================================================================
__global__ __launch_bounds__(128)
void ln_kernel(const float* __restrict__ X, const float* __restrict__ g,
               const float* __restrict__ b, unsigned short* __restrict__ out)
{
    __shared__ float s1[4], s2[4];
    const int row = blockIdx.x;
    const int t   = threadIdx.x;
    const float v = X[(size_t)row * D_ + t];
    float a = v, q = v * v;
    #pragma unroll
    for (int off = 16; off; off >>= 1) {
        a += __shfl_xor(a, off, 32);
        q += __shfl_xor(q, off, 32);
    }
    if ((t & 31) == 0) { s1[t >> 5] = a; s2[t >> 5] = q; }
    __syncthreads();
    const float mean = (s1[0] + s1[1] + s1[2] + s1[3]) * (1.0f / D_);
    const float msq  = (s2[0] + s2[1] + s2[2] + s2[3]) * (1.0f / D_);
    const float var  = msq - mean * mean;
    const float r    = rsqrtf(var + 1e-5f);
    out[(size_t)row * D_ + t] = f2bf_u((v - mean) * r * g[t] + b[t]);
}

// ---------------- prep kernels ----------------
__global__ void mask_kernel(const float* __restrict__ x, float* __restrict__ mask_idx)
{
    const int n = blockIdx.x * blockDim.x + threadIdx.x;
    if (n >= N_) return;
    float s = 0.f;
    for (int i = 0; i < DIN_; ++i) s += x[(size_t)n * DIN_ + i];
    mask_idx[n] = (s != 0.f) ? 1.f : 0.f;
}

__global__ void zero_deg_kernel(int* __restrict__ deg)
{
    const int n = blockIdx.x * blockDim.x + threadIdx.x;
    if (n < N_) deg[n] = 0;
}

__global__ void deg_kernel(const int* __restrict__ edge_index, int* __restrict__ deg)
{
    const int e = blockIdx.x * blockDim.x + threadIdx.x;
    if (e >= E_) return;
    const int dst = edge_index[E_ + e];          // edge_index[1][e]
    atomicAdd(&deg[dst], 1);
}

__global__ void nodeproj_kernel(const float* __restrict__ x,
                                const float* __restrict__ node_W,
                                const float* __restrict__ node_b,
                                const float* __restrict__ deg_emb,
                                const int* __restrict__ deg,
                                float* __restrict__ h)
{
    const int tid = blockIdx.x * blockDim.x + threadIdx.x;
    const int n = tid / D_, d = tid % D_;
    float acc = node_b[d];
    #pragma unroll 8
    for (int i = 0; i < DIN_; ++i)
        acc = fmaf(x[(size_t)n * DIN_ + i], node_W[(size_t)i * D_ + d], acc);
    int dg = deg[n];
    dg = (dg < 0) ? 0 : ((dg > N_ - 1) ? N_ - 1 : dg);
    h[(size_t)n * D_ + d] = acc + deg_emb[(size_t)dg * D_ + d];
}

__global__ void edge_wmean_kernel(const float* __restrict__ edge_W,
                                  const float* __restrict__ edge_b,
                                  float* __restrict__ wmean)
{
    const int s = threadIdx.x;
    if (s < S_) {
        float m = 0.f;
        for (int k = 0; k < S_; ++k) m += edge_W[s * S_ + k];
        wmean[s] = m * (1.0f / S_);
    }
    if (s == 0) {
        float m = 0.f;
        for (int k = 0; k < S_; ++k) m += edge_b[k];
        wmean[S_] = m * (1.0f / S_);
    }
}

// bias[i,j] = sp_emb[spl[i,j]] + dot(edge_feat[i,j,:], wmean) + bmean; masked -> -1e30
__global__ void biasprep_kernel(const int* __restrict__ spl,
                                const float* __restrict__ edge_feat,
                                const float* __restrict__ sp_emb,
                                const float* __restrict__ wmean,
                                const float* __restrict__ mask_idx,
                                float* __restrict__ biasM)
{
    const size_t tid = (size_t)blockIdx.x * blockDim.x + threadIdx.x;
    const int i = (int)(tid / N_), j = (int)(tid % N_);
    const float* ef = edge_feat + tid * S_;
    float em = wmean[S_];
    #pragma unroll
    for (int s = 0; s < S_; ++s) em = fmaf(ef[s], wmean[s], em);
    const float sp = sp_emb[spl[tid]];
    const float mk = mask_idx[i] * mask_idx[j];
    biasM[tid] = (mk == 0.f) ? -1.0e30f : (sp + em);
}

__global__ void cast_bf16_kernel(const float* __restrict__ in, unsigned short* __restrict__ out)
{
    const size_t tid = (size_t)blockIdx.x * blockDim.x + threadIdx.x;
    out[tid] = f2bf_u(in[tid]);
}

// ======================================================================
extern "C" void kernel_launch(void* const* d_in, const int* in_sizes, int n_in,
                              void* d_out, int out_size, void* d_ws, size_t ws_size,
                              hipStream_t stream)
{
    (void)in_sizes; (void)n_in; (void)out_size; (void)ws_size;

    const float* x         = (const float*)d_in[0];
    const int*   edge_idx  = (const int*)  d_in[1];
    const int*   spl       = (const int*)  d_in[2];
    const float* edge_feat = (const float*)d_in[3];
    const float* node_W    = (const float*)d_in[4];
    const float* node_b    = (const float*)d_in[5];
    const float* deg_emb   = (const float*)d_in[6];
    const float* sp_emb    = (const float*)d_in[7];
    const float* edge_W    = (const float*)d_in[8];
    const float* edge_b    = (const float*)d_in[9];
    const float* ln1_g     = (const float*)d_in[10];
    const float* ln1_b     = (const float*)d_in[11];
    const float* Wq        = (const float*)d_in[12];
    const float* bq        = (const float*)d_in[13];
    const float* Wk        = (const float*)d_in[14];
    const float* bk        = (const float*)d_in[15];
    const float* Wv        = (const float*)d_in[16];
    const float* bv        = (const float*)d_in[17];
    const float* Wo        = (const float*)d_in[18];
    const float* bo        = (const float*)d_in[19];
    const float* ln2_g     = (const float*)d_in[20];
    const float* ln2_b     = (const float*)d_in[21];
    const float* ff_W      = (const float*)d_in[22];
    const float* ff_b      = (const float*)d_in[23];
    const float* out_W     = (const float*)d_in[24];
    const float* out_b     = (const float*)d_in[25];

    // -------- workspace carve --------
    char* w = (char*)d_ws;
    auto alloc = [&](size_t bytes) { char* p = w; w += (bytes + 255) & ~(size_t)255; return p; };
    float*          mask_idx = (float*)         alloc((size_t)N_ * 4);
    int*            deg      = (int*)           alloc((size_t)N_ * 4);
    float*          wmean    = (float*)         alloc(8 * 4);
    float*          h        = (float*)         alloc((size_t)N_ * D_ * 4);
    float*          xout     = (float*)         alloc((size_t)N_ * D_ * 4);
    float*          biasM    = (float*)         alloc((size_t)N_ * N_ * 4);
    float*          logits   = (float*)         alloc((size_t)N_ * N_ * 4);
    unsigned short* P        = (unsigned short*)alloc((size_t)N_ * N_ * 2);
    unsigned short* xln      = (unsigned short*)alloc((size_t)N_ * D_ * 2);
    unsigned short* Qb       = (unsigned short*)alloc((size_t)N_ * D_ * 2);
    unsigned short* Kb       = (unsigned short*)alloc((size_t)N_ * D_ * 2);
    unsigned short* Vb       = (unsigned short*)alloc((size_t)N_ * D_ * 2);
    unsigned short* cat      = (unsigned short*)alloc((size_t)N_ * H_ * D_ * 2);
    unsigned short* hb       = (unsigned short*)alloc((size_t)N_ * D_ * 2);

    const float scale = 0.088388347648f;   // 1/sqrt(128)

    // -------- prep --------
    mask_kernel    <<<(N_ + 255) / 256, 256, 0, stream>>>(x, mask_idx);
    zero_deg_kernel<<<(N_ + 255) / 256, 256, 0, stream>>>(deg);
    deg_kernel     <<<(E_ + 255) / 256, 256, 0, stream>>>(edge_idx, deg);
    nodeproj_kernel<<<(N_ * D_) / 256, 256, 0, stream>>>(x, node_W, node_b, deg_emb, deg, h);
    edge_wmean_kernel<<<1, 32, 0, stream>>>(edge_W, edge_b, wmean);
    biasprep_kernel<<<(int)(((size_t)N_ * N_) / 256), 256, 0, stream>>>(
        spl, edge_feat, sp_emb, wmean, mask_idx, biasM);

    const dim3 gTile(N_ / 16, D_ / 64);     // 192 x 2 strips (16 x 64 each)
    const dim3 gAttn(N_ / 16, N_ / 64);     // 192 x 48 strips

    for (int l = 0; l < L_; ++l) {
        ln_kernel<<<N_, 128, 0, stream>>>(h, ln1_g + l * D_, ln1_b + l * D_, xln);

        for (int hd = 0; hd < H_; ++hd) {
            const size_t wofs = (size_t)(l * H_ + hd) * D_ * D_;
            const size_t bofs = (size_t)(l * H_ + hd) * D_;
            // Q/K/V projections: bf16 WMMA, f32 weights converted in the loader
            wmma_gemm_kernel<false, true, true, 4><<<gTile, 32, 0, stream>>>(
                xln, D_, Wq + wofs, D_, bq + bofs, nullptr, 0, nullptr, Qb, D_, 0, D_);
            wmma_gemm_kernel<false, true, true, 4><<<gTile, 32, 0, stream>>>(
                xln, D_, Wk + wofs, D_, bk + bofs, nullptr, 0, nullptr, Kb, D_, 0, D_);
            wmma_gemm_kernel<false, true, true, 4><<<gTile, 32, 0, stream>>>(
                xln, D_, Wv + wofs, D_, bv + bofs, nullptr, 0, nullptr, Vb, D_, 0, D_);

            wmma_qkt_kernel<<<gAttn, 32, 0, stream>>>(Qb, Kb, biasM, logits, scale);
            softmax_kernel <<<N_, 256, 0, stream>>>(logits, P);

            // o = P @ V into cat[:, hd*D : (hd+1)*D]  (bf16)
            wmma_gemm_kernel<false, false, true, 4><<<gTile, 32, 0, stream>>>(
                P, N_, Vb, D_, nullptr, nullptr, 0, nullptr, cat, H_ * D_, hd * D_, N_);
        }

        // x_out = cat @ Wo[l] + bo[l] + h
        wmma_gemm_kernel<false, true, false, 4><<<gTile, 32, 0, stream>>>(
            cat, H_ * D_, Wo + (size_t)l * H_ * D_ * D_, D_, bo + l * D_,
            h, D_, xout, nullptr, D_, 0, H_ * D_);

        // h = LN2(x_out) @ ff_W[l] + ff_b[l] + x_out
        ln_kernel<<<N_, 128, 0, stream>>>(xout, ln2_g + l * D_, ln2_b + l * D_, xln);
        wmma_gemm_kernel<false, true, false, 4><<<gTile, 32, 0, stream>>>(
            xln, D_, ff_W + (size_t)l * D_ * D_, D_, ff_b + l * D_,
            xout, D_, h, nullptr, D_, 0, D_);
    }

    // out = h @ out_W + out_b
    cast_bf16_kernel<<<(N_ * D_) / 256, 256, 0, stream>>>(h, hb);
    wmma_gemm_kernel<false, true, false, 4><<<gTile, 32, 0, stream>>>(
        hb, D_, out_W, D_, out_b, nullptr, 0, (float*)d_out, nullptr, D_, 0, D_);
}
